// AutoEPMoELayer_73023033966714
// MI455X (gfx1250) — matrix operand order repaired and checked by reading, not compiled, not run on hardware
//
#include <hip/hip_runtime.h>
#include <hip/hip_bf16.h>

// ---------------- problem constants (fixed by the reference) ----------------
#define HDIM   1024
#define FDIM   4096
#define NEXP   8
#define NTOK   8192          // B*S = 4*2048
#define CAP    8192          // worst-case rows per expert
#define NPAIR  (2 * NTOK)    // token-expert pairs (top-2)

typedef __attribute__((ext_vector_type(16))) __bf16 v16bf;
typedef __attribute__((ext_vector_type(8)))  float  v8f;

// pack two fp32 into one dword of bf16x2 (lowers to v_cvt_pk_bf16_f32)
__device__ __forceinline__ unsigned pk_bf16(float lo, float hi) {
    unsigned short a = __builtin_bit_cast(unsigned short, (__bf16)lo);
    unsigned short b = __builtin_bit_cast(unsigned short, (__bf16)hi);
    return (unsigned)a | ((unsigned)b << 16);
}

// ---------------------------------------------------------------------------
// Kernel 0: zero the per-expert counters (workspace is not re-poisoned).
// ---------------------------------------------------------------------------
__global__ void moe_zero_cnt(int* cnt) {
    if (threadIdx.x < NEXP) cnt[threadIdx.x] = 0;
}

// ---------------------------------------------------------------------------
// Kernel 1: routing. One wave32 per token: 8 logits, top-2, renormalized
// softmax weights, append pair-id (2t+k) to the chosen experts' lists.
// Output is deterministic: list order may vary, but every result is written
// to a pair-indexed location and per-row math is order-independent.
// ---------------------------------------------------------------------------
__global__ __launch_bounds__(256)
void moe_route(const float* __restrict__ x,
               const float* __restrict__ gate_w,
               int* __restrict__ cnt,
               int* __restrict__ tok,
               float* __restrict__ wgt) {
    const int t    = (blockIdx.x * blockDim.x + threadIdx.x) >> 5;  // token
    const int lane = threadIdx.x & 31;
    if (t >= NTOK) return;

    const float* xr = x + (size_t)t * HDIM;
    float acc[NEXP];
#pragma unroll
    for (int e = 0; e < NEXP; ++e) acc[e] = 0.f;

    for (int h = lane; h < HDIM; h += 32) {
        const float xv = xr[h];
#pragma unroll
        for (int e = 0; e < NEXP; ++e)
            acc[e] += xv * gate_w[e * HDIM + h];
    }
#pragma unroll
    for (int e = 0; e < NEXP; ++e) {
#pragma unroll
        for (int off = 16; off > 0; off >>= 1)
            acc[e] += __shfl_xor(acc[e], off, 32);
    }

    if (lane == 0) {
        int e0 = 0;
#pragma unroll
        for (int e = 1; e < NEXP; ++e) if (acc[e] > acc[e0]) e0 = e;
        int e1 = (e0 == 0) ? 1 : 0;
#pragma unroll
        for (int e = 0; e < NEXP; ++e)
            if (e != e0 && acc[e] > acc[e1]) e1 = e;
        // renormalized top-2 softmax: w0 = exp(l0)/(exp(l0)+exp(l1))
        const float w0 = 1.f / (1.f + __expf(acc[e1] - acc[e0]));
        const float w1 = 1.f - w0;

        int i0 = atomicAdd(&cnt[e0], 1);
        tok[e0 * CAP + i0] = 2 * t + 0;
        wgt[e0 * CAP + i0] = w0;
        int i1 = atomicAdd(&cnt[e1], 1);
        tok[e1 * CAP + i1] = 2 * t + 1;
        wgt[e1 * CAP + i1] = w1;
    }
}

// ---------------------------------------------------------------------------
// Kernel 2: grouped GEMM1, fused gate: h[p, n] = silu(x@w1)*(x@w3) in bf16.
// Block tile 64(M) x 128(N), BK=32, 8 waves; wave = 16x64 strip -> 4 wmma
// tiles per B matrix per K-step, A fragment reused 8x.  Double-buffered LDS:
// one __syncthreads per K-step; global loads of tile k+1 overlap WMMA on k.
// ---------------------------------------------------------------------------
__global__ __launch_bounds__(256)
void moe_ffn1(const float* __restrict__ x,
              const float* __restrict__ w1,
              const float* __restrict__ w3,
              const int* __restrict__ cnt,
              const int* __restrict__ tok,
              __bf16* __restrict__ hbuf) {
    const int e     = blockIdx.z;
    const int n_e   = cnt[e];
    const int mbase = blockIdx.y * 64;
    if (mbase >= n_e) return;                // empty tile for this expert
    const int nbase = blockIdx.x * 128;

    __shared__ __bf16 As [2][64][32];        // 8 KB
    __shared__ __bf16 B1t[2][128][32];       // 16 KB  (transposed: [n][k])
    __shared__ __bf16 B3t[2][128][32];       // 16 KB
    __shared__ int    rowp[64];

    const int tid = threadIdx.x;
    if (tid < 64) {
        const int pi = mbase + tid;
        rowp[tid] = (pi < n_e) ? tok[e * CAP + pi] : -1;
    }
    __syncthreads();

    const float* w1e = w1 + (size_t)e * HDIM * FDIM;
    const float* w3e = w3 + (size_t)e * HDIM * FDIM;

    // loader mappings (256 threads)
    const int arow  = tid >> 2;              // 0..63
    const int akseg = (tid & 3) * 8;         // 0,8,16,24
    const int bkp   = (tid & 15) * 2;        // even k: 0..30
    const int bnseg = (tid >> 4) * 8;        // 0..120

    const int p_a = rowp[arow];
    const float* aptr = (p_a >= 0) ? (x + (size_t)(p_a >> 1) * HDIM + akseg)
                                   : x;      // dummy, masked below

    // wave mapping
    const int wv = tid >> 5, lane = tid & 31;
    const int m0 = (wv & 3) * 16;
    const int n0 = (wv >> 2) * 64;
    const int hl = lane >> 4, ll = lane & 15;

    v8f acc1[4] = {}, acc3[4] = {};

    // staging registers
    float a_s[8], s1a[8], s1b[8], s3a[8], s3b[8];

    auto load_stage = [&](int k0) {
        if (p_a >= 0) {
            __builtin_prefetch(aptr + k0 + 32, 0, 1);
#pragma unroll
            for (int j = 0; j < 8; ++j) a_s[j] = aptr[k0 + j];
        } else {
#pragma unroll
            for (int j = 0; j < 8; ++j) a_s[j] = 0.f;
        }
        const float* p1 = w1e + (size_t)(k0 + bkp) * FDIM + nbase + bnseg;
        const float* p3 = w3e + (size_t)(k0 + bkp) * FDIM + nbase + bnseg;
        __builtin_prefetch(p1 + 32 * FDIM, 0, 1);
        __builtin_prefetch(p3 + 32 * FDIM, 0, 1);
#pragma unroll
        for (int j = 0; j < 8; ++j) {
            s1a[j] = p1[j];  s1b[j] = p1[FDIM + j];
            s3a[j] = p3[j];  s3b[j] = p3[FDIM + j];
        }
    };

    auto store_stage = [&](int buf) {
        uint4 av;
        av.x = pk_bf16(a_s[0], a_s[1]);
        av.y = pk_bf16(a_s[2], a_s[3]);
        av.z = pk_bf16(a_s[4], a_s[5]);
        av.w = pk_bf16(a_s[6], a_s[7]);
        *(uint4*)&As[buf][arow][akseg] = av;        // one ds_store_b128
#pragma unroll
        for (int j = 0; j < 8; ++j) {               // 8 ds_store_b32 each
            *(unsigned*)&B1t[buf][bnseg + j][bkp] = pk_bf16(s1a[j], s1b[j]);
            *(unsigned*)&B3t[buf][bnseg + j][bkp] = pk_bf16(s3a[j], s3b[j]);
        }
    };

    load_stage(0);
    store_stage(0);
    __syncthreads();

    const int NK = HDIM / 32;
    for (int kk = 0; kk < NK; ++kk) {
        const int cur = kk & 1;
        const bool more = (kk + 1 < NK);
        if (more) load_stage((kk + 1) * 32);

        // A fragment (16x32 bf16, contiguous k = 16*half + i, matching B)
        v16bf a;
        {
            const __bf16* ar = &As[cur][m0 + ll][hl * 16];
#pragma unroll
            for (int i = 0; i < 8; ++i) { a[i] = ar[i]; a[i + 8] = ar[i + 8]; }
        }
#pragma unroll
        for (int c = 0; c < 4; ++c) {
            v16bf b1, b3;
            const __bf16* q1 = &B1t[cur][n0 + c * 16 + ll][hl * 16];
            const __bf16* q3 = &B3t[cur][n0 + c * 16 + ll][hl * 16];
#pragma unroll
            for (int i = 0; i < 16; ++i) { b1[i] = q1[i]; b3[i] = q3[i]; }
            acc1[c] = __builtin_amdgcn_wmma_f32_16x16x32_bf16(
                false, a, false, b1, (short)0, acc1[c], false, false);
            acc3[c] = __builtin_amdgcn_wmma_f32_16x16x32_bf16(
                false, a, false, b3, (short)0, acc3[c], false, false);
        }

        if (more) store_stage(cur ^ 1);
        __syncthreads();
    }

    // epilogue: h = silu(a1) * a3, store bf16 at pair row p
#pragma unroll
    for (int c = 0; c < 4; ++c) {
#pragma unroll
        for (int r = 0; r < 8; ++r) {
            const int mrow = m0 + r + 8 * hl;     // C layout: M = r + 8*half
            const int p = rowp[mrow];
            if (p >= 0) {
                const float a1v = acc1[c][r];
                const float a3v = acc3[c][r];
                const float s = (a1v / (1.f + __expf(-a1v))) * a3v;
                hbuf[(size_t)p * FDIM + nbase + n0 + c * 16 + ll] = (__bf16)s;
            }
        }
    }
}

// ---------------------------------------------------------------------------
// Kernel 3: grouped GEMM2: part[p, n] = wgt[p] * (h[p,:] @ w2[e][:,n]).
// Same tiling (64x128, BK=32, double-buffered); A already bf16 (pure copy).
// ---------------------------------------------------------------------------
__global__ __launch_bounds__(256)
void moe_ffn2(const __bf16* __restrict__ hbuf,
              const float* __restrict__ w2,
              const int* __restrict__ cnt,
              const int* __restrict__ tok,
              const float* __restrict__ wgt,
              float* __restrict__ part) {
    const int e     = blockIdx.z;
    const int n_e   = cnt[e];
    const int mbase = blockIdx.y * 64;
    if (mbase >= n_e) return;
    const int nbase = blockIdx.x * 128;

    __shared__ __bf16 As[2][64][32];         // 8 KB
    __shared__ __bf16 Bt[2][128][32];        // 16 KB (transposed: [n][k])
    __shared__ int    rowp[64];
    __shared__ float  roww[64];

    const int tid = threadIdx.x;
    if (tid < 64) {
        const int pi = mbase + tid;
        const bool v = (pi < n_e);
        rowp[tid] = v ? tok[e * CAP + pi] : -1;
        roww[tid] = v ? wgt[e * CAP + pi] : 0.f;
    }
    __syncthreads();

    const float* w2e = w2 + (size_t)e * FDIM * HDIM;

    const int arow  = tid >> 2;
    const int akseg = (tid & 3) * 8;
    const int bkp   = (tid & 15) * 2;
    const int bnseg = (tid >> 4) * 8;

    const int p_a = rowp[arow];
    const __bf16* aptr = (p_a >= 0) ? (hbuf + (size_t)p_a * FDIM + akseg)
                                    : hbuf;

    const int wv = tid >> 5, lane = tid & 31;
    const int m0 = (wv & 3) * 16;
    const int n0 = (wv >> 2) * 64;
    const int hl = lane >> 4, ll = lane & 15;

    v8f acc[4] = {};

    uint4 a4;                                  // 8 bf16 staged as dwords
    float s2a[8], s2b[8];

    auto load_stage = [&](int k0) {
        if (p_a >= 0) {
            __builtin_prefetch(aptr + k0 + 32, 0, 1);
            a4 = *(const uint4*)(aptr + k0);
        } else {
            a4 = make_uint4(0u, 0u, 0u, 0u);
        }
        const float* p2 = w2e + (size_t)(k0 + bkp) * HDIM + nbase + bnseg;
        __builtin_prefetch(p2 + 32 * HDIM, 0, 1);
#pragma unroll
        for (int j = 0; j < 8; ++j) {
            s2a[j] = p2[j];  s2b[j] = p2[HDIM + j];
        }
    };

    auto store_stage = [&](int buf) {
        *(uint4*)&As[buf][arow][akseg] = a4;   // one ds_store_b128
#pragma unroll
        for (int j = 0; j < 8; ++j)
            *(unsigned*)&Bt[buf][bnseg + j][bkp] = pk_bf16(s2a[j], s2b[j]);
    };

    load_stage(0);
    store_stage(0);
    __syncthreads();

    const int NK = FDIM / 32;
    for (int kk = 0; kk < NK; ++kk) {
        const int cur = kk & 1;
        const bool more = (kk + 1 < NK);
        if (more) load_stage((kk + 1) * 32);

        v16bf a;
        {
            const __bf16* ar = &As[cur][m0 + ll][hl * 16];
#pragma unroll
            for (int i = 0; i < 8; ++i) { a[i] = ar[i]; a[i + 8] = ar[i + 8]; }
        }
#pragma unroll
        for (int c = 0; c < 4; ++c) {
            v16bf b;
            const __bf16* br = &Bt[cur][n0 + c * 16 + ll][hl * 16];
#pragma unroll
            for (int i = 0; i < 16; ++i) b[i] = br[i];
            acc[c] = __builtin_amdgcn_wmma_f32_16x16x32_bf16(
                false, a, false, b, (short)0, acc[c], false, false);
        }

        if (more) store_stage(cur ^ 1);
        __syncthreads();
    }

    // epilogue: scale by router weight, write partial (no atomics)
#pragma unroll
    for (int c = 0; c < 4; ++c) {
#pragma unroll
        for (int r = 0; r < 8; ++r) {
            const int mrow = m0 + r + 8 * hl;
            const int p = rowp[mrow];
            if (p >= 0) {
                part[(size_t)p * HDIM + nbase + n0 + c * 16 + ll] =
                    roww[mrow] * acc[c][r];
            }
        }
    }
}

// ---------------------------------------------------------------------------
// Kernel 4: combine the two expert contributions per token.
// ---------------------------------------------------------------------------
__global__ void moe_combine(const float* __restrict__ part,
                            float* __restrict__ out) {
    const size_t i = (size_t)blockIdx.x * blockDim.x + threadIdx.x;
    if (i >= (size_t)NTOK * HDIM) return;
    const size_t t = i / HDIM;
    const size_t j = i - t * HDIM;
    out[i] = part[t * 2 * HDIM + j] + part[t * 2 * HDIM + HDIM + j];
}

// ---------------------------------------------------------------------------
// Host launcher
// ---------------------------------------------------------------------------
extern "C" void kernel_launch(void* const* d_in, const int* in_sizes, int n_in,
                              void* d_out, int out_size, void* d_ws, size_t ws_size,
                              hipStream_t stream) {
    const float* x      = (const float*)d_in[0];   // [4,2048,1024]
    const float* gate_w = (const float*)d_in[1];   // [8,1024]
    const float* w1     = (const float*)d_in[2];   // [8,1024,4096]
    const float* w2     = (const float*)d_in[3];   // [8,4096,1024]
    const float* w3     = (const float*)d_in[4];   // [8,1024,4096]
    float* out = (float*)d_out;

    // workspace layout
    char* ws = (char*)d_ws;
    const size_t off_cnt  = 0;
    const size_t off_tok  = 256;
    const size_t off_wgt  = off_tok + (size_t)NEXP * CAP * sizeof(int);
    const size_t off_h    = off_wgt + (size_t)NEXP * CAP * sizeof(float);
    const size_t off_part = off_h   + (size_t)NPAIR * FDIM * sizeof(__bf16);
    int*    cnt  = (int*)(ws + off_cnt);
    int*    tok  = (int*)(ws + off_tok);
    float*  wgt  = (float*)(ws + off_wgt);
    __bf16* hbuf = (__bf16*)(ws + off_h);
    float*  part = (float*)(ws + off_part);

    moe_zero_cnt<<<1, 32, 0, stream>>>(cnt);

    moe_route<<<NTOK / 8, 256, 0, stream>>>(x, gate_w, cnt, tok, wgt);

    dim3 g1(FDIM / 128, NTOK / 64, NEXP);   // (32,128,8) — empty tiles exit early
    moe_ffn1<<<g1, 256, 0, stream>>>(x, w1, w3, cnt, tok, hbuf);

    dim3 g2(HDIM / 128, NTOK / 64, NEXP);   // (8,128,8)
    moe_ffn2<<<g2, 256, 0, stream>>>(hbuf, w2, cnt, tok, wgt, part);

    moe_combine<<<((size_t)NTOK * HDIM + 255) / 256, 256, 0, stream>>>(part, out);
}
